// GraphEncoder_28217935134813
// MI455X (gfx1250) — compile-verified, matrix operand order checked by reading, not run on hardware
//
#include <hip/hip_runtime.h>

// ---------------------------------------------------------------------------
// Problem dims (from reference): B=16, LD=1024, N=64, L=32, IN=1024, H=1024
// ---------------------------------------------------------------------------
#define DB   16
#define DLD  1024
#define DN   64
#define DL   32
#define DIN  1024
#define DH   1024

typedef __attribute__((ext_vector_type(16))) __bf16 v16bf;
typedef __attribute__((ext_vector_type(8)))  __bf16 v8bf;
typedef __attribute__((ext_vector_type(8)))  float  v8f;

#define GEMM_RELU 1
#define GEMM_ACC  2

// ---------------------------------------------------------------------------
// WMMA bf16 fragment read from an LDS tile (rows x 32, row-major, bf16).
// A fragment layout (16-bit A 16x32, wave32):
//   lanes 0-15: M=lane,    VGPR0..3 = K 0..7,   VGPR4..7 = K16..23
//   lanes16-31: M=lane-16, VGPR0..3 = K 8..15,  VGPR4..7 = K24..31
// B fragment is symmetric over columns; tiles are staged as Bt[N][K] slabs so
// the same reader works for both operands.
// ---------------------------------------------------------------------------
__device__ __forceinline__ v16bf frag_lds(const __bf16* base, int row, int lane) {
  const __bf16* p = base + (row + (lane & 15)) * 32 + (lane >> 4) * 8;
  v8bf lo = *(const v8bf*)(p);
  v8bf hp = *(const v8bf*)(p + 16);
  return __builtin_shufflevector(lo, hp, 0,1,2,3,4,5,6,7,8,9,10,11,12,13,14,15);
}

// ---------------------------------------------------------------------------
// Generic batched GEMM:  C[M,N] = op(A[M,K] @ B[K,N]),  B given transposed
// (Bt[N,K] row-major). 256 threads = 8 waves. Block tile 128x128 staged in
// LDS (8KB A + 8KB B per 32-wide K slab, next slab double-buffered through
// registers). Each wave owns a 32x64 tile = 2x4 v_wmma_f32_16x16x32_bf16
// accumulators (f32). flags: RELU then optional per-column scale (matches
// relu(xW)*v), ACC adds into existing f32 C. Optional bf16 mirror output Cb.
// Edge tiles handled by clamped loads + guarded stores (K must be mult of 32).
// ---------------------------------------------------------------------------
__global__ __launch_bounds__(256)
void gemm_wmma_bf16(const __bf16* __restrict__ A, long strideA,
                    const __bf16* __restrict__ Bt, long strideB,
                    float* C, long strideC,
                    __bf16* Cb, long strideCb,
                    int M, int N, int K,
                    int lda, int ldb, int ldc,
                    const float* __restrict__ colscale, int flags) {
  __shared__ __align__(16) __bf16 As[128 * 32];   // 8KB
  __shared__ __align__(16) __bf16 Bs[128 * 32];   // 8KB

  int batch = blockIdx.z;
  A  += (long)batch * strideA;
  Bt += (long)batch * strideB;
  if (C)  C  += (long)batch * strideC;
  if (Cb) Cb += (long)batch * strideCb;

  int tid  = threadIdx.x;
  int lane = tid & 31;
  int w    = tid >> 5;          // 0..7
  int wm   = w >> 1;            // 0..3 -> 32-row strips
  int wn   = w & 1;             // 0..1 -> 64-col strips
  int mBlk = blockIdx.y * 128;
  int nBlk = blockIdx.x * 128;
  int mW   = wm * 32;
  int nW   = wn * 64;

  // cooperative staging: 512 16B-chunks per tile, 2 chunks/thread/tile
  // chunk c: row = c>>2 (0..127), 16B column = c&3
  int c0 = tid * 2, c1 = tid * 2 + 1;

  auto gload = [&](const __bf16* base, int ld, int rBase, int maxR, int k0,
                   int c) -> uint4 {
    int row = rBase + (c >> 2);
    if (row >= maxR) row = maxR - 1;             // clamp: rows never stored
    return *(const uint4*)(base + (long)row * ld + k0 + (c & 3) * 8);
  };
  auto sstore = [&](uint4 va0, uint4 va1, uint4 vb0, uint4 vb1) {
    *(uint4*)(As + (c0 >> 2) * 32 + (c0 & 3) * 8) = va0;
    *(uint4*)(As + (c1 >> 2) * 32 + (c1 & 3) * 8) = va1;
    *(uint4*)(Bs + (c0 >> 2) * 32 + (c0 & 3) * 8) = vb0;
    *(uint4*)(Bs + (c1 >> 2) * 32 + (c1 & 3) * 8) = vb1;
  };

  // prime first K-slab
  {
    uint4 ra0 = gload(A,  lda, mBlk, M, 0, c0);
    uint4 ra1 = gload(A,  lda, mBlk, M, 0, c1);
    uint4 rb0 = gload(Bt, ldb, nBlk, N, 0, c0);
    uint4 rb1 = gload(Bt, ldb, nBlk, N, 0, c1);
    sstore(ra0, ra1, rb0, rb1);
  }
  __syncthreads();

  v8f acc[2][4] = {};

  for (int k0 = 0; k0 < K; k0 += 32) {
    bool more = (k0 + 32) < K;
    uint4 na0, na1, nb0, nb1;
    if (more) {                                   // register double-buffer
      na0 = gload(A,  lda, mBlk, M, k0 + 32, c0);
      na1 = gload(A,  lda, mBlk, M, k0 + 32, c1);
      nb0 = gload(Bt, ldb, nBlk, N, k0 + 32, c0);
      nb1 = gload(Bt, ldb, nBlk, N, k0 + 32, c1);
      if (k0 + 64 < K) {                          // hint slab after next
        __builtin_prefetch(A  + (long)(mBlk + (c0 >> 2)) * lda + k0 + 64, 0, 3);
        __builtin_prefetch(Bt + (long)(nBlk + (c0 >> 2)) * ldb + k0 + 64, 0, 3);
      }
    }

    v16bf af[2], bfr[4];
    af[0]  = frag_lds(As, mW,      lane);
    af[1]  = frag_lds(As, mW + 16, lane);
#pragma unroll
    for (int nj = 0; nj < 4; ++nj) bfr[nj] = frag_lds(Bs, nW + nj * 16, lane);

#pragma unroll
    for (int mi = 0; mi < 2; ++mi)
#pragma unroll
      for (int nj = 0; nj < 4; ++nj)
        acc[mi][nj] = __builtin_amdgcn_wmma_f32_16x16x32_bf16(
            false, af[mi], false, bfr[nj], (short)0, acc[mi][nj], false, false);

    __syncthreads();                              // all reads done
    if (more) {
      sstore(na0, na1, nb0, nb1);
      __syncthreads();                            // stores visible
    }
  }

  int hi = lane >> 4;
  int nl = lane & 15;
#pragma unroll
  for (int mi = 0; mi < 2; ++mi)
#pragma unroll
    for (int nj = 0; nj < 4; ++nj)
#pragma unroll
      for (int v = 0; v < 8; ++v) {
        int m = mBlk + mW + mi * 16 + hi * 8 + v;  // C/D layout: VGPRv -> M=v(+8)
        int n = nBlk + nW + nj * 16 + nl;          // lane&15 -> N
        if (m < M && n < N) {
          float val = acc[mi][nj][v];
          if (flags & GEMM_RELU) val = fmaxf(val, 0.0f);
          if (colscale) val *= colscale[n];
          long off = (long)m * ldc + n;
          if (C) {
            float o = val;
            if (flags & GEMM_ACC) o += C[off];
            C[off] = o;
          }
          if (Cb) Cb[off] = (__bf16)val;
        }
      }
}

// ---------------------------------------------------------------------------
// Small helper kernels
// ---------------------------------------------------------------------------
__global__ void k_convert(const float* __restrict__ src, __bf16* __restrict__ dst, long n) {
  long i = (long)blockIdx.x * blockDim.x + threadIdx.x;
  if (i < n) dst[i] = (__bf16)src[i];
}

// dst[C,R] = (bf16) src[R,C]^T  (weights -> Bt form)
__global__ void k_convert_T(const float* __restrict__ src, __bf16* __restrict__ dst,
                            int R, int Ccols, long n) {
  long i = (long)blockIdx.x * blockDim.x + threadIdx.x;
  if (i >= n) return;
  int r = (int)(i / Ccols), c = (int)(i % Ccols);
  dst[(long)c * R + r] = (__bf16)src[i];
}

// gb = (bf16)( (float)ga * v[col] ), row length H
__global__ void k_colscale_bf16(const __bf16* __restrict__ ga, const float* __restrict__ v,
                                __bf16* __restrict__ gb, long n) {
  long i = (long)blockIdx.x * blockDim.x + threadIdx.x;
  if (i < n) gb[i] = (__bf16)((float)ga[i] * v[i & (DH - 1)]);
}

// last valid doc hidden per sample -> bf16 row
__global__ void k_lastdoc(const float* __restrict__ doc, const int* __restrict__ mask,
                          __bf16* __restrict__ lastdoc) {
  int b = blockIdx.x, tid = threadIdx.x;
  __shared__ int cnt;
  if (tid == 0) cnt = 0;
  __syncthreads();
  int c = 0;
  for (int i = tid; i < DLD; i += 256) c += (mask[b * DLD + i] == 0) ? 1 : 0;
  atomicAdd(&cnt, c);
  __syncthreads();
  int row = cnt - 1;
  for (int h = tid; h < DH; h += 256)
    lastdoc[b * DH + h] = (__bf16)doc[((long)b * DLD + row) * DH + h];
}

// score2 (dot of r1v with r2n rows), softmax over L=32, weighted-sum attn rows
__global__ void k_score2_nodeinit(const float* __restrict__ r1v,
                                  const __bf16* __restrict__ r2n,
                                  const __bf16* __restrict__ attn_bf,
                                  const unsigned char* __restrict__ nmask,
                                  float* __restrict__ node_f,
                                  __bf16* __restrict__ node_b) {
  int p = blockIdx.x;                 // 0..B*N-1
  int b = p >> 6;
  int tid = threadIdx.x, lane = tid & 31, w = tid >> 5;
  __shared__ float sc[DL];
  __shared__ float e[DL];
  const float* rv = r1v + b * DH;
  for (int li = 0; li < 4; ++li) {
    int l = w * 4 + li;
    const __bf16* row = r2n + ((long)p * DL + l) * DH;
    float s = 0.0f;
    for (int h = lane; h < DH; h += 32) s += rv[h] * (float)row[h];
    for (int o = 16; o > 0; o >>= 1) s += __shfl_xor(s, o, 32);
    if (lane == 0) sc[l] = s;
  }
  __syncthreads();
  if (tid < 32) {
    float v = sc[tid];
    if (nmask[p * DL + tid]) v = -3.0e38f;
    float m = v;
    for (int o = 16; o > 0; o >>= 1) m = fmaxf(m, __shfl_xor(m, o, 32));
    float ex = __expf(v - m);
    float sum = ex;
    for (int o = 16; o > 0; o >>= 1) sum += __shfl_xor(sum, o, 32);
    e[tid] = ex / sum;
  }
  __syncthreads();
  for (int h = tid; h < DH; h += 256) {
    float acc = 0.0f;
    const __bf16* base = attn_bf + (long)p * DL * DH + h;
    for (int l = 0; l < DL; ++l) acc += e[l] * (float)base[(long)l * DH];
    node_f[(long)p * DH + h] = acc;
    node_b[(long)p * DH + h] = (__bf16)acc;
  }
}

// one GAT round: mask scores with edge==0 -> -1e5, softmax over j, mix rows
__global__ void k_round(const float* __restrict__ S, const int* __restrict__ edge,
                        const float* __restrict__ node_old,
                        float* __restrict__ node_new, __bf16* __restrict__ node_new_b) {
  int p = blockIdx.x;                 // b*N+i
  int b = p >> 6, i = p & (DN - 1);
  int tid = threadIdx.x;
  __shared__ float ea[DN];
  __shared__ float inv;
  if (tid < DN) {
    float v = S[(long)b * DN * DN + i * DN + tid];
    if (edge[(long)b * DN * DN + i * DN + tid] == 0) v = -100000.0f;
    ea[tid] = v;
  }
  __syncthreads();
  if (tid == 0) {
    float m = ea[0];
    for (int j = 1; j < DN; ++j) m = fmaxf(m, ea[j]);
    float s = 0.0f;
    for (int j = 0; j < DN; ++j) { ea[j] = __expf(ea[j] - m); s += ea[j]; }
    inv = 1.0f / s;
  }
  __syncthreads();
  float scl = inv;
  for (int h = tid; h < DH; h += 256) {
    float acc = 0.0f;
    for (int j = 0; j < DN; ++j)
      acc += ea[j] * node_old[((long)b * DN + j) * DH + h];
    acc *= scl;
    node_new[(long)p * DH + h] = acc;
    node_new_b[(long)p * DH + h] = (__bf16)acc;
  }
}

__global__ void k_finalmask(const int* __restrict__ edge, int* __restrict__ fmask) {
  int i = blockIdx.x * blockDim.x + threadIdx.x;   // b*N+n
  if (i >= DB * DN) return;
  int s = 0;
  for (int j = 0; j < DN; ++j) s += edge[(long)i * DN + j];
  fmask[i] = (s == 0) ? 1 : 0;
}

// softmax over N=64 nodes for each (b, ld); -inf where masked
__global__ void k_final_softmax(const float* __restrict__ score, const int* __restrict__ fmask,
                                __bf16* __restrict__ alpha) {
  int idx = blockIdx.x;               // b*LD+ld
  int b = idx >> 10;
  int tid = threadIdx.x;              // 64 threads
  __shared__ float v[DN];
  __shared__ float inv;
  float x = score[(long)idx * DN + tid];
  if (fmask[b * DN + tid]) x = -3.0e38f;
  v[tid] = x;
  __syncthreads();
  if (tid == 0) {
    float m = v[0];
    for (int j = 1; j < DN; ++j) m = fmaxf(m, v[j]);
    float s = 0.0f;
    for (int j = 0; j < DN; ++j) { v[j] = __expf(v[j] - m); s += v[j]; }
    inv = 1.0f / s;
  }
  __syncthreads();
  alpha[(long)idx * DN + tid] = (__bf16)(v[tid] * inv);
}

// outT[b][h][n] = node[b][n][h]  (bf16), for the alpha @ output GEMM
__global__ void k_transpose_out(const __bf16* __restrict__ node_b, __bf16* __restrict__ outT) {
  long i = (long)blockIdx.x * blockDim.x + threadIdx.x;
  if (i >= (long)DB * DN * DH) return;
  int b = (int)(i >> 16);             // / (N*H)
  int rem = (int)(i & 65535);
  int n = rem >> 10, h = rem & (DH - 1);
  outT[(long)b * DH * DN + (long)h * DN + n] = node_b[i];
}

// ---------------------------------------------------------------------------
// Host launcher
// ---------------------------------------------------------------------------
extern "C" void kernel_launch(void* const* d_in, const int* in_sizes, int n_in,
                              void* d_out, int out_size, void* d_ws, size_t ws_size,
                              hipStream_t stream) {
  (void)in_sizes; (void)n_in; (void)out_size; (void)ws_size;

  const float* doc   = (const float*)d_in[0];
  const float* nemb  = (const float*)d_in[1];
  const float* wnode = (const float*)d_in[2];
  const float* u3    = (const float*)d_in[3];
  const float* u4    = (const float*)d_in[4];
  const float* W2    = (const float*)d_in[5];
  const float* v2    = (const float*)d_in[6];
  const float* W1    = (const float*)d_in[7];
  const float* v1    = (const float*)d_in[8];
  const float* W0    = (const float*)d_in[9];
  const float* v0    = (const float*)d_in[10];
  const int*   dmask = (const int*)d_in[11];
  const unsigned char* nmask = (const unsigned char*)d_in[12];
  const int*   edge  = (const int*)d_in[13];
  float* out = (float*)d_out;

  // ---- workspace carving --------------------------------------------------
  char* ws = (char*)d_ws;
  size_t off = 0;
  auto carve = [&](size_t bytes) { size_t o = off; off += (bytes + 255) & ~(size_t)255; return o; };
  const size_t WB = (size_t)DH * DH * sizeof(__bf16);     // 2MB per transposed weight

  __bf16* wnT   = (__bf16*)(ws + carve(WB));
  __bf16* w2T   = (__bf16*)(ws + carve(WB));
  __bf16* w1T   = (__bf16*)(ws + carve(WB));
  __bf16* w0T   = (__bf16*)(ws + carve(WB));
  __bf16* u3T   = (__bf16*)(ws + carve(WB));
  __bf16* u4T   = (__bf16*)(ws + carve(WB));
  __bf16* lastd = (__bf16*)(ws + carve((size_t)DB * DH * 2));
  float*  r1v   = (float*)(ws + carve((size_t)DB * DH * 4));
  float*  nodeAf = (float*)(ws + carve((size_t)DB * DN * DH * 4));
  __bf16* nodeAb = (__bf16*)(ws + carve((size_t)DB * DN * DH * 2));
  float*  nodeBf = (float*)(ws + carve((size_t)DB * DN * DH * 4));
  __bf16* nodeBb = (__bf16*)(ws + carve((size_t)DB * DN * DH * 2));
  __bf16* ga    = (__bf16*)(ws + carve((size_t)DB * DN * DH * 2));
  __bf16* gb    = (__bf16*)(ws + carve((size_t)DB * DN * DH * 2));
  float*  Smat  = (float*)(ws + carve((size_t)DB * DN * DN * 4));
  int*    fmask = (int*)(ws + carve((size_t)DB * DN * 4));
  float*  score = (float*)(ws + carve((size_t)DB * DLD * DN * 4));
  __bf16* alpha = (__bf16*)(ws + carve((size_t)DB * DLD * DN * 2));
  __bf16* outT  = (__bf16*)(ws + carve((size_t)DB * DH * DN * 2));
  char*   BIGA  = ws + carve((size_t)64 << 20);           // 64MB reuse region
  char*   BIGB  = ws + carve((size_t)64 << 20);           // 64MB reuse region

  __bf16* node_bf = (__bf16*)BIGA;                        // [32768,1024]
  __bf16* attn_bf = (__bf16*)BIGB;                        // [32768,1024]
  __bf16* r2n     = (__bf16*)BIGA;                        // reuse after attn done
  __bf16* doc_bf  = (__bf16*)BIGA;                        // [16384,1024]  (32MB)
  __bf16* r1d     = (__bf16*)(BIGA + ((size_t)32 << 20)); // [16384,1024]  (32MB)
  __bf16* matched = (__bf16*)BIGB;                        // [16384,1024]  (32MB)

  const long MKN = (long)DH * DH;                         // 1M elements
  dim3 blk(256);

  // ---- 1. weight transposes to bf16 ---------------------------------------
  k_convert_T<<<dim3((MKN + 255) / 256), blk, 0, stream>>>(wnode, wnT, DIN, DH, MKN);
  k_convert_T<<<dim3((MKN + 255) / 256), blk, 0, stream>>>(W2, w2T, DH, DH, MKN);
  k_convert_T<<<dim3((MKN + 255) / 256), blk, 0, stream>>>(W1, w1T, DH, DH, MKN);
  k_convert_T<<<dim3((MKN + 255) / 256), blk, 0, stream>>>(W0, w0T, DH, DH, MKN);
  k_convert_T<<<dim3((MKN + 255) / 256), blk, 0, stream>>>(u3, u3T, DH, DH, MKN);
  k_convert_T<<<dim3((MKN + 255) / 256), blk, 0, stream>>>(u4, u4T, DH, DH, MKN);

  // ---- 2. node_emb -> bf16 ------------------------------------------------
  const long NENB = (long)DB * DN * DL * DIN;             // 33,554,432
  k_convert<<<dim3((unsigned)((NENB + 255) / 256)), blk, 0, stream>>>(nemb, node_bf, NENB);

  // ---- 3. attn_node = node_emb @ weightnode  (bf16 out) -------------------
  gemm_wmma_bf16<<<dim3(DH / 128, (DB * DN * DL) / 128, 1), blk, 0, stream>>>(
      node_bf, 0, wnT, 0, nullptr, 0, attn_bf, 0,
      DB * DN * DL, DH, DIN, DIN, DIN, DH, nullptr, 0);

  // ---- 4. last_doc, r1v = relu(last_doc @ W2) * v2 ------------------------
  k_lastdoc<<<dim3(DB), blk, 0, stream>>>(doc, dmask, lastd);
  gemm_wmma_bf16<<<dim3(DH / 128, 1, 1), blk, 0, stream>>>(
      lastd, 0, w2T, 0, r1v, 0, nullptr, 0,
      DB, DH, DH, DH, DH, DH, v2, GEMM_RELU);

  // ---- 5. r2n = relu(attn_node @ W2)  (bf16) ------------------------------
  gemm_wmma_bf16<<<dim3(DH / 128, (DB * DN * DL) / 128, 1), blk, 0, stream>>>(
      attn_bf, 0, w2T, 0, nullptr, 0, r2n, 0,
      DB * DN * DL, DH, DH, DH, DH, DH, nullptr, GEMM_RELU);

  // ---- 6. softmax over L + node_initial -----------------------------------
  k_score2_nodeinit<<<dim3(DB * DN), blk, 0, stream>>>(r1v, r2n, attn_bf, nmask,
                                                       nodeAf, nodeAb);

  // ---- 7. three GAT rounds ------------------------------------------------
  for (int r = 0; r < 3; ++r) {
    const float*  curF = (r & 1) ? nodeBf : nodeAf;
    const __bf16* curB = (r & 1) ? nodeBb : nodeAb;
    float*  nxtF = (r & 1) ? nodeAf : nodeBf;
    __bf16* nxtB = (r & 1) ? nodeAb : nodeBb;

    gemm_wmma_bf16<<<dim3(DH / 128, (DB * DN) / 128, 1), blk, 0, stream>>>(
        curB, 0, w1T, 0, nullptr, 0, ga, 0,
        DB * DN, DH, DH, DH, DH, DH, nullptr, GEMM_RELU);
    k_colscale_bf16<<<dim3((unsigned)((MKN + 255) / 256)), blk, 0, stream>>>(ga, v1, gb, MKN);
    gemm_wmma_bf16<<<dim3(1, 1, DB), blk, 0, stream>>>(
        ga, (long)DN * DH, gb, (long)DN * DH, Smat, (long)DN * DN, nullptr, 0,
        DN, DN, DH, DH, DH, DN, nullptr, 0);
    k_round<<<dim3(DB * DN), blk, 0, stream>>>(Smat, edge, curF, nxtF, nxtB);
  }
  // after rounds 0,1,2: final node state lives in nodeB*

  // ---- 8. doc -> bf16, r1d = relu(doc @ W0), r2o = relu(out @ W0)*v0 ------
  const long DOCN = (long)DB * DLD * DH;                  // 16,777,216
  k_convert<<<dim3((unsigned)((DOCN + 255) / 256)), blk, 0, stream>>>(doc, doc_bf, DOCN);
  gemm_wmma_bf16<<<dim3(DH / 128, (DB * DLD) / 128, 1), blk, 0, stream>>>(
      doc_bf, 0, w0T, 0, nullptr, 0, r1d, 0,
      DB * DLD, DH, DH, DH, DH, DH, nullptr, GEMM_RELU);
  gemm_wmma_bf16<<<dim3(DH / 128, (DB * DN) / 128, 1), blk, 0, stream>>>(
      nodeBb, 0, w0T, 0, nullptr, 0, ga /*reuse as r2o*/, 0,
      DB * DN, DH, DH, DH, DH, DH, v0, GEMM_RELU);

  // ---- 9. score[b] = r1d_b @ r2o_b^T  (batched) ---------------------------
  gemm_wmma_bf16<<<dim3(1, DLD / 128, DB), blk, 0, stream>>>(
      r1d, (long)DLD * DH, ga, (long)DN * DH, score, (long)DLD * DN, nullptr, 0,
      DLD, DN, DH, DH, DH, DN, nullptr, 0);

  // ---- 10. final softmax over nodes, alpha (bf16) -------------------------
  k_finalmask<<<dim3((DB * DN + 255) / 256), blk, 0, stream>>>(edge, fmask);
  k_final_softmax<<<dim3(DB * DLD), dim3(DN), 0, stream>>>(score, fmask, alpha);

  // ---- 11. matched[b] = alpha_b @ output_b  (batched, K=64) ---------------
  k_transpose_out<<<dim3((unsigned)(((long)DB * DN * DH + 255) / 256)), blk, 0, stream>>>(
      nodeBb, outT);
  gemm_wmma_bf16<<<dim3(DH / 128, DLD / 128, DB), blk, 0, stream>>>(
      alpha, (long)DLD * DN, outT, (long)DH * DN, nullptr, 0,
      matched, (long)DLD * DH,
      DLD, DH, DN, DN, DN, DH, nullptr, 0);

  // ---- 12. out = matched @ update3 + doc @ update4 ------------------------
  gemm_wmma_bf16<<<dim3(DH / 128, (DB * DLD) / 128, 1), blk, 0, stream>>>(
      matched, 0, u3T, 0, out, 0, nullptr, 0,
      DB * DLD, DH, DH, DH, DH, DH, nullptr, 0);
  gemm_wmma_bf16<<<dim3(DH / 128, (DB * DLD) / 128, 1), blk, 0, stream>>>(
      doc_bf, 0, u4T, 0, out, 0, nullptr, 0,
      DB * DLD, DH, DH, DH, DH, DH, nullptr, GEMM_ACC);
}
